// ConvSelfAttention_45913200394581
// MI455X (gfx1250) — compile-verified, matrix operand order checked by reading, not run on hardware
//
#include <hip/hip_runtime.h>

typedef __attribute__((ext_vector_type(16))) _Float16 v16h;
typedef __attribute__((ext_vector_type(8)))  _Float16 v8h;
typedef __attribute__((ext_vector_type(8)))  float    v8f;

constexpr int kB   = 4;
constexpr int kT   = 32;
constexpr int kHS  = 19;
constexpr int kWS  = 19;
constexpr int kC   = 64;
constexpr int kNH  = 4;
constexpr int kD   = 16;
constexpr int kNPIX = kHS * kWS;        // 361
constexpr int kIMG  = kB * kT;          // 128
constexpr int kPADW = 21;               // zero-padded 21x21
constexpr int kPADE = kPADW * kPADW * kC; // 28224 halves

// d_ws layout:
//   [0,       73728)  : wT  f16 [9 taps][64 cout][64 cin]   (kv conv weights, transposed)
//   [73728,   ...  )  : kv  f32 [128 img][361 pix][64 ch]
constexpr int kWT_BYTES = 9 * 64 * 64 * 2;   // 73728, 256B-aligned

// ---------------------------------------------------------------------------
// Kernel 0: transpose + f16-convert the fused (all-heads) kv conv weights.
// w_kv: [NH][3][3][C][D] fp32  ->  wT: [tap][n=h*16+d][cin] f16
// ---------------------------------------------------------------------------
__global__ void prep_weights(const float* __restrict__ w_kv,
                             _Float16* __restrict__ wT) {
  int i = blockIdx.x * blockDim.x + threadIdx.x;
  if (i >= 9 * 64 * 64) return;
  int tap = i >> 12;
  int n   = (i >> 6) & 63;
  int ci  = i & 63;
  int h = n >> 4, d = n & 15;
  int ky = tap / 3, kx = tap % 3;
  int src = (((h * 3 + ky) * 3 + kx) * kC + ci) * kD + d;
  wT[i] = (_Float16)w_kv[src];
}

// ---------------------------------------------------------------------------
// Kernel 1: fused kv conv for all 4 heads as one implicit GEMM per image.
// Per image: [361 x 576] x [576 x 64], 16x16 tile per wave, K=32 WMMA chunks.
// ---------------------------------------------------------------------------
__global__ __launch_bounds__(256) void kv_conv(const float* __restrict__ x,
                                               const _Float16* __restrict__ wT,
                                               const float* __restrict__ b_kv,
                                               float* __restrict__ kv) {
  __shared__ __align__(16) _Float16 sx[kPADE];
  const int img = blockIdx.x;
  const int tid = threadIdx.x;

  // zero-pad borders, then fill interior with f16-converted x
  for (int i = tid; i < kPADE; i += 256) sx[i] = (_Float16)0.f;
  __syncthreads();
  const float* xi = x + (long)img * (kNPIX * kC);
  for (int i = tid; i < kNPIX * kC; i += 256) {
    int p = i >> 6, c = i & 63;
    int py = p / kWS, px = p % kWS;
    sx[((py + 1) * kPADW + (px + 1)) * kC + c] = (_Float16)xi[i];
  }
  __syncthreads();

  const int wv = tid >> 5, lane = tid & 31;
  const int lh = lane & 15;      // M row (A) / N col (B,C,D)
  const int lg = lane >> 4;      // K-striping group per ISA 16-bit layout

  // 23 M-tiles x 4 N-tiles = 92 tiles, round-robined over 8 waves
  for (int tile = wv; tile < 92; tile += 8) {
    int tm = tile % 23, tn = tile / 23;
    int mp = tm * 16 + lh;
    int mc = mp < kNPIX ? mp : kNPIX - 1;   // clamp loads; stores are guarded
    int py = mc / kWS, px = mc % kWS;
    int n = tn * 16 + lh;

    v8f acc = {};
#pragma unroll
    for (int tap = 0; tap < 9; ++tap) {
      int ky = tap / 3, kx = tap % 3;
      int abase = ((py + ky) * kPADW + (px + kx)) * kC + lg * 8;
      int bbase = tap * 4096 + n * 64 + lg * 8;
#pragma unroll
      for (int c0 = 0; c0 < 64; c0 += 32) {
        // elements e<8: K = c0 + lg*8 + e ; e>=8: K = c0 + 16 + lg*8 + (e-8)
        v8h alo = *(const v8h*)(sx + abase + c0);
        v8h ahi = *(const v8h*)(sx + abase + c0 + 16);
        v8h blo = *(const v8h*)(wT + bbase + c0);
        v8h bhi = *(const v8h*)(wT + bbase + c0 + 16);
        v16h a, b;
#pragma unroll
        for (int e = 0; e < 8; ++e) {
          a[e] = alo[e]; a[e + 8] = ahi[e];
          b[e] = blo[e]; b[e + 8] = bhi[e];
        }
        acc = __builtin_amdgcn_wmma_f32_16x16x32_f16(
            false, a, false, b, (short)0, acc, false, false);
      }
    }

    float bias = b_kv[n];  // b_kv flat [NH*D] == channel n
    float* kvo = kv + (long)img * (kNPIX * kC);
#pragma unroll
    for (int v = 0; v < 8; ++v) {
      int M = tm * 16 + lg * 8 + v;     // C/D layout: VGPR v -> row v (+8 hi lanes)
      if (M < kNPIX) kvo[M * kC + n] = acc[v] + bias;
    }
  }
}

// ---------------------------------------------------------------------------
// Kernel 2: att_k mini-conv (64->4ch), softmax over key time (q-branch and
// b_att cancel exactly in the tk-softmax), weighted kv sum, broadcast over tq.
// One block per (b, pixel); wave h owns head h, lane == key time t.
// ---------------------------------------------------------------------------
__global__ __launch_bounds__(128) void attn_softmax_out(
    const float* __restrict__ kv, const float* __restrict__ w_att,
    float* __restrict__ out) {
  __shared__ float s_soft[kNH * kT];
  __shared__ float s_part[2 * kC];
  const int bp  = blockIdx.x;
  const int b   = bp / kNPIX, pix = bp % kNPIX;
  const int py  = pix / kWS, px = pix % kWS;
  const int tid = threadIdx.x;
  const int t   = tid & 31, h = tid >> 5;

  // att_k[b,t,pix,h]: 3x3 conv of kv head-h channels with key-half of w_att
  float a = 0.f;
#pragma unroll
  for (int ky = 0; ky < 3; ++ky) {
    int ny = py + ky - 1;
#pragma unroll
    for (int kx = 0; kx < 3; ++kx) {
      int nx = px + kx - 1;
      if (ny < 0 || ny >= kHS || nx < 0 || nx >= kWS) continue;
      const float* kp = kv + ((long)(b * kT + t) * kNPIX + ny * kWS + nx) * kC + h * kD;
      const float* wp = w_att + ((h * 3 + ky) * 3 + kx) * (2 * kD);  // first D = key part
#pragma unroll
      for (int d = 0; d < kD; ++d) a += kp[d] * wp[d];
    }
  }

  // softmax over key time: wave32 butterfly (lane == t)
  float m = a;
  for (int off = 16; off > 0; off >>= 1) m = fmaxf(m, __shfl_xor(m, off));
  float e = __expf(a - m);
  float s = e;
  for (int off = 16; off > 0; off >>= 1) s += __shfl_xor(s, off);
  s_soft[h * kT + t] = e / s;
  __syncthreads();

  // out_base[c] = sum_t soft[t, c/16] * kv[t, pix, c]; split t over two halves
  const int c = tid & 63, halfid = tid >> 6;
  float acc = 0.f;
#pragma unroll
  for (int i = 0; i < 16; ++i) {
    int tk = halfid * 16 + i;
    acc += s_soft[(c >> 4) * kT + tk] *
           kv[((long)(b * kT + tk) * kNPIX + pix) * kC + c];
  }
  s_part[halfid * kC + c] = acc;
  __syncthreads();

  if (tid < kC) {
    float val = s_part[tid] + s_part[kC + tid];
    // output is tq-invariant: broadcast across all 32 query times
    float* op = out + ((long)b * kT * kNPIX + pix) * kC + tid;
    for (int tq = 0; tq < kT; ++tq) op[(long)tq * kNPIX * kC] = val;
  }
}

// ---------------------------------------------------------------------------
extern "C" void kernel_launch(void* const* d_in, const int* in_sizes, int n_in,
                              void* d_out, int out_size, void* d_ws, size_t ws_size,
                              hipStream_t stream) {
  const float* x     = (const float*)d_in[0];
  // d_in[1] = w_q, d_in[2] = b_q: provably unused (cancel in tk-softmax)
  const float* w_kv  = (const float*)d_in[3];
  const float* b_kv  = (const float*)d_in[4];
  const float* w_att = (const float*)d_in[5];
  // d_in[6] = b_att: constant along tk, cancels in softmax

  _Float16* wT    = (_Float16*)d_ws;
  float*    kvbuf = (float*)((char*)d_ws + kWT_BYTES);

  prep_weights<<<(9 * 64 * 64 + 255) / 256, 256, 0, stream>>>(w_kv, wT);
  kv_conv<<<kIMG, 256, 0, stream>>>(x, wT, b_kv, kvbuf);
  attn_softmax_out<<<kB * kNPIX, 128, 0, stream>>>(kvbuf, w_att, (float*)d_out);
}
